// ScaledDotProductAttention_24739011625884
// MI455X (gfx1250) — matrix operand order repaired and checked by reading, not compiled, not run on hardware
//
#include <hip/hip_runtime.h>

typedef __attribute__((ext_vector_type(16))) _Float16 v16h;
typedef __attribute__((ext_vector_type(8)))  _Float16 h8;
typedef __attribute__((ext_vector_type(4)))  _Float16 h4;
typedef __attribute__((ext_vector_type(8)))  float    v8f;

#define WMMA_F16(a, b, c) \
  __builtin_amdgcn_wmma_f32_16x16x32_f16(false, (a), false, (b), (short)0, (c), false, false)

// exp(score/8): fold scale (and log2e when hardware exp2 is directly reachable) into Q.
#if __has_builtin(__builtin_amdgcn_exp2f)
  #define QSCALE 0.18033688011112042f   // 0.125 * log2(e)
  #define EXPFN(x) __builtin_amdgcn_exp2f(x)
#else
  #define QSCALE 0.125f
  #define EXPFN(x) __expf(x)
#endif

constexpr int Bb = 8;      // batch
constexpr int Ls = 2048;   // sequence length
constexpr int Dd = 64;     // head dim
constexpr int TQ = 64;     // query rows per block (4 waves x 16)
constexpr int TK = 64;     // kv rows per tile
constexpr int PITCH = 72;  // LDS row pitch in halves: 144 B = 36 dwords -> conflict-free b128 lanes

union HV { v16h v; h8 q[2]; _Float16 h[16]; };

__global__ __launch_bounds__(128)
void ScaledDotProductAttention_fa_kernel(const float* __restrict__ Q,
                                         const float* __restrict__ K,
                                         const float* __restrict__ V,
                                         const float* __restrict__ qmask,
                                         const float* __restrict__ kmask,
                                         float* __restrict__ O)
{
    // LDS: Kld[64][72] f16 | Vt[64][72] f16 (Vt[d][krow]) | P scratch 4x[16][72] f16
    __shared__ _Float16 lds_[3 * 64 * PITCH];
    _Float16* Kld = lds_;
    _Float16* Vt  = lds_ + 64 * PITCH;

    const int tid  = threadIdx.x;
    const int wave = tid >> 5;
    const int lane = tid & 31;
    const int n    = lane & 15;   // N index (column of C/D, row of A)
    const int g    = lane >> 4;   // lane group
    _Float16* Pw   = lds_ + 2 * 64 * PITCH + wave * 16 * PITCH;

    const int blk   = blockIdx.x;
    const int b     = blk >> 5;            // / (Ls/TQ) = /32
    const int q0    = (blk & 31) * TQ;
    const int qrow0 = q0 + wave * 16;

    // ---------------- Load Q tile into A-layout, f32 -> f16, scale folded in ----------
    // A-layout: row M = lane%16; half h of group g holds K=(h%8)+8g+(h<8?0:16) (+32c)
    HV aQ[2];
    {
        const float* qrow = Q + (b * Ls + qrow0 + n) * Dd;
        #pragma unroll
        for (int c = 0; c < 2; ++c) {
            #pragma unroll
            for (int part = 0; part < 2; ++part) {
                const int d0 = 32 * c + part * 16 + 8 * g;
                const float4 f0 = *(const float4*)(qrow + d0);
                const float4 f1 = *(const float4*)(qrow + d0 + 4);
                aQ[c].h[part * 8 + 0] = (_Float16)(f0.x * QSCALE);
                aQ[c].h[part * 8 + 1] = (_Float16)(f0.y * QSCALE);
                aQ[c].h[part * 8 + 2] = (_Float16)(f0.z * QSCALE);
                aQ[c].h[part * 8 + 3] = (_Float16)(f0.w * QSCALE);
                aQ[c].h[part * 8 + 4] = (_Float16)(f1.x * QSCALE);
                aQ[c].h[part * 8 + 5] = (_Float16)(f1.y * QSCALE);
                aQ[c].h[part * 8 + 6] = (_Float16)(f1.z * QSCALE);
                aQ[c].h[part * 8 + 7] = (_Float16)(f1.w * QSCALE);
            }
        }
    }

    v8f acc[4];
    #pragma unroll
    for (int dt = 0; dt < 4; ++dt) acc[dt] = {};
    float rsum[8] = {0.f, 0.f, 0.f, 0.f, 0.f, 0.f, 0.f, 0.f};

    // ---------------- Main loop over KV tiles ----------------
    for (int it = 0; it < Ls / TK; ++it) {
        const int kv0 = it * TK;
        __syncthreads();   // all waves done reading previous Kld/Vt

        const float4* K4 = (const float4*)(K + (b * Ls + kv0) * Dd);
        const float4* V4 = (const float4*)(V + (b * Ls + kv0) * Dd);

        // K: row-major staging, contiguous stores (conflict-free)
        #pragma unroll
        for (int i = 0; i < 8; ++i) {
            const int idx4 = tid + i * 128;         // 1024 float4 per tile
            const int r  = idx4 >> 4;               // kv row 0..63
            const int cc = (idx4 & 15) << 2;        // d column 0..60
            const float4 kf = K4[idx4];
            h4 kh = { (_Float16)kf.x, (_Float16)kf.y, (_Float16)kf.z, (_Float16)kf.w };
            *(h4*)(Kld + r * PITCH + cc) = kh;
        }
        // V: transposed staging; lanes walk the kv-row (fast index of Vt) so the
        // b16 scatter stores have 2-byte inter-lane stride (no 16-way bank conflicts)
        #pragma unroll
        for (int i = 0; i < 8; ++i) {
            const int r    = (tid & 15) + 16 * (i & 3);       // kv row 0..63
            const int col4 = (tid >> 4) + 8 * (i >> 2);       // d/4  0..15
            const float4 vf = V4[r * 16 + col4];
            Vt[(col4 * 4 + 0) * PITCH + r] = (_Float16)vf.x;
            Vt[(col4 * 4 + 1) * PITCH + r] = (_Float16)vf.y;
            Vt[(col4 * 4 + 2) * PITCH + r] = (_Float16)vf.z;
            Vt[(col4 * 4 + 3) * PITCH + r] = (_Float16)vf.w;
        }
        if (it + 1 < Ls / TK) {   // prefetch next 16 KB tile (128 B per thread)
            __builtin_prefetch((const char*)(K4 + 1024) + tid * 128, 0, 1);
            __builtin_prefetch((const char*)(V4 + 1024) + tid * 128, 0, 1);
        }
        __syncthreads();

        // --------- S = Q Kt (four 16x16 N-tiles), P = exp2(S)*km ---------
        #pragma unroll
        for (int nt = 0; nt < 4; ++nt) {
            // B-layout: N = lane%16; half h of group g holds K=(h%8)+8g+(h<8?0:16) (+32c)
            const _Float16* kr = Kld + (nt * 16 + n) * PITCH;
            HV bK0, bK1;
            bK0.q[0] = *(const h8*)(kr + 8 * g);
            bK0.q[1] = *(const h8*)(kr + 16 + 8 * g);
            bK1.q[0] = *(const h8*)(kr + 32 + 8 * g);
            bK1.q[1] = *(const h8*)(kr + 32 + 16 + 8 * g);

            v8f s = {};
            s = WMMA_F16(aQ[0].v, bK0.v, s);
            s = WMMA_F16(aQ[1].v, bK1.v, s);

            const float kmv = kmask[b * Ls + kv0 + nt * 16 + n];
            _Float16* pdst = Pw + nt * 16 + n;   // column k = nt*16+n, row stride PITCH
            #pragma unroll
            for (int v = 0; v < 8; ++v) {
                const float p = EXPFN(s[v]) * kmv;
                rsum[v] += p;
                pdst[(v + 8 * g) * PITCH] = (_Float16)p;   // row m = v + 8g
            }
        }
        asm volatile("s_wait_dscnt 0" ::: "memory");

        // --------- Re-read P in A-layout ---------
        HV aP[2];
        {
            const _Float16* pr = Pw + n * PITCH;
            #pragma unroll
            for (int c = 0; c < 2; ++c) {
                aP[c].q[0] = *(const h8*)(pr + 32 * c + 8 * g);
                aP[c].q[1] = *(const h8*)(pr + 32 * c + 16 + 8 * g);
            }
        }

        // --------- O += P * V (four 16-wide d tiles) ---------
        #pragma unroll
        for (int dt = 0; dt < 4; ++dt) {
            const _Float16* vr = Vt + (dt * 16 + n) * PITCH;   // Vt[d][krow]
            HV bV0, bV1;
            bV0.q[0] = *(const h8*)(vr + 8 * g);
            bV0.q[1] = *(const h8*)(vr + 16 + 8 * g);
            bV1.q[0] = *(const h8*)(vr + 32 + 8 * g);
            bV1.q[1] = *(const h8*)(vr + 32 + 16 + 8 * g);
            acc[dt] = WMMA_F16(aP[0].v, bV0.v, acc[dt]);
            acc[dt] = WMMA_F16(aP[1].v, bV1.v, acc[dt]);
        }
    }

    // ---------------- Epilogue: normalize & store ----------------
    float inv[8];
    #pragma unroll
    for (int v = 0; v < 8; ++v) {
        float r = rsum[v];
        r += __shfl_xor(r, 1, 32);
        r += __shfl_xor(r, 2, 32);
        r += __shfl_xor(r, 4, 32);
        r += __shfl_xor(r, 8, 32);   // sum over the 16 N-lanes (stays in half-wave)
        const float qv = qmask[b * Ls + qrow0 + v + 8 * g];
        inv[v] = (qv != 0.f) ? (1.f / fmaxf(r, 2e-15f)) : 0.f;
    }
    #pragma unroll
    for (int dt = 0; dt < 4; ++dt)
        #pragma unroll
        for (int v = 0; v < 8; ++v)
            O[(b * Ls + qrow0 + v + 8 * g) * Dd + dt * 16 + n] = acc[dt][v] * inv[v];
}

extern "C" void kernel_launch(void* const* d_in, const int* in_sizes, int n_in,
                              void* d_out, int out_size, void* d_ws, size_t ws_size,
                              hipStream_t stream) {
    const float* Q  = (const float*)d_in[0];
    const float* K  = (const float*)d_in[1];
    const float* V  = (const float*)d_in[2];
    const float* qm = (const float*)d_in[3];
    const float* km = (const float*)d_in[4];
    float* O = (float*)d_out;
    dim3 grid(Bb * (Ls / TQ));   // 8 * 32 = 256 blocks
    dim3 block(128);             // 4 waves (wave32)
    hipLaunchKernelGGL(ScaledDotProductAttention_fa_kernel, grid, block, 0, stream,
                       Q, K, V, qm, km, O);
}
